// SequentialModel_3367254360541
// MI455X (gfx1250) — matrix operand order repaired and checked by reading, not compiled, not run on hardware
//
#include <hip/hip_runtime.h>
#include <hip/hip_bf16.h>
#include <math.h>

// GPT-2 forward (B=1, T=2048, NX=1024, NH=16, HD=64, NL=4) for gfx1250.
// Dense math via v_wmma_f32_16x16x32_bf16 (fp32 accum); GEMM A-tiles staged
// to LDS with the Tensor Data Mover (tensor_load_to_lds + s_wait_tensorcnt).

#define Tt  2048
#define NXc 1024
#define NHc 16
#define HDc 64
#define NLc 4

typedef __bf16 bf16;
typedef __attribute__((ext_vector_type(16))) __bf16 v16bf;
typedef __attribute__((ext_vector_type(8)))  __bf16 v8bf;
typedef __attribute__((ext_vector_type(8)))  float  v8f;
typedef __attribute__((ext_vector_type(4)))  unsigned int u32x4;
typedef __attribute__((ext_vector_type(8)))  int i32x8;
typedef __attribute__((ext_vector_type(4)))  int i32x4;

// Load a 16-element bf16 fragment as two 16B chunks (chunk1 at +off2 elements).
__device__ __forceinline__ v16bf frag_ld(const bf16* p, int off2) {
  v8bf lo = *(const v8bf*)(p);
  v8bf hi = *(const v8bf*)(p + off2);
  return __builtin_shufflevector(lo, hi, 0,1,2,3,4,5,6,7,8,9,10,11,12,13,14,15);
}

__device__ __forceinline__ v8f wmma_bf16(v16bf a, v16bf b, v8f c) {
  // 8 args: (neg_a, A, neg_b, B, c_mod, C, reuse_a, reuse_b)
  return __builtin_amdgcn_wmma_f32_16x16x32_bf16(false, a, false, b, (short)0, c, false, false);
}

__device__ __forceinline__ float gelu_new_f(float x) {
  const float c = 0.79788456080286535588f; // sqrt(2/pi)
  return 0.5f * x * (1.0f + tanhf(c * (x + 0.044715f * x * x * x)));
}

enum { EPI_BF16 = 0, EPI_GELU_BF16 = 1, EPI_RES_F32 = 2, EPI_F32 = 3 };

// C[M,N] = A[M,K] @ W[K,N] + bias (+ epilogue). A,W bf16 row-major.
// 256 threads = 8 waves; block tile 128x128; wave computes 16 rows x 128 cols.
// A tile (128x32 bf16, padded to 40-elem rows) staged via TDM.
template <int EPI>
__global__ void __launch_bounds__(256)
k_gemm(const bf16* __restrict__ A, const bf16* __restrict__ W,
       const float* __restrict__ bias, const float* __restrict__ R,
       float* __restrict__ Cf, bf16* __restrict__ Cb,
       int M, int N, int K) {
  __shared__ bf16 Alds[128][40]; // [m][k], 80B row stride (TDM pad: 16dw data + 4dw pad)
  __shared__ bf16 Blds[128][40]; // [n][k]

  const int bn = blockIdx.x * 128;
  const int bm = blockIdx.y * 128;
  const int t  = threadIdx.x;
  const int wv = t >> 5, lane = t & 31;
  const int h  = lane >> 4, nlo = lane & 15;
  const bool wave0 = (__builtin_amdgcn_readfirstlane((int)threadIdx.x) >> 5) == 0;

  v8f acc[8] = {};

  const int bn0 = (t & 15) * 8;       // 0..120
  const int bk0 = (t >> 4) * 2;       // 0..30

  for (int k0 = 0; k0 < K; k0 += 32) {
    // --- Stage A 128x32 via Tensor Data Mover (one descriptor, wave 0 only) ---
    if (wave0) {
      const unsigned long long ga =
          (unsigned long long)(uintptr_t)(A + (size_t)bm * K + k0);
      const unsigned int la = (unsigned int)(uintptr_t)&Alds[0][0]; // LDS offset = addr[31:0]
      u32x4 g0;
      g0[0] = 1u;                                   // count=1, user desc, no gather
      g0[1] = la;                                   // lds_addr
      g0[2] = (unsigned int)ga;                     // global_addr[31:0]
      g0[3] = (unsigned int)(ga >> 32) | (2u << 30);// global_addr[56:32] | type=2
      i32x8 g1;
      g1[0] = (int)((1u << 16)   // data_size=1 (2B elements)
                  | (1u << 20)   // pad_enable
                  | (3u << 22)   // pad_interval code 3 -> every 16 DWORDs (one 64B row)
                  | (3u << 25)); // pad_amount  code 3 -> 4 DWORDs (8 elems) pad
      g1[1] = (int)(((unsigned)K & 0xFFFFu) << 16);                 // tensor_dim0[15:0]
      g1[2] = (int)((((unsigned)K >> 16) & 0xFFFFu) |
                    (((unsigned)M & 0xFFFFu) << 16));               // dim0[31:16] | dim1[15:0]
      g1[3] = (int)((((unsigned)M >> 16) & 0xFFFFu) | (32u << 16)); // dim1[31:16] | tile_dim0=32
      g1[4] = 128;                                                  // tile_dim1=128, tile_dim2=0
      g1[5] = K;                                                    // tensor_dim0_stride[31:0]
      g1[6] = 0;                                                    // stride hi | dim1_stride lo
      g1[7] = 0;
      i32x4 gz4 = {0, 0, 0, 0};
      i32x8 gz8 = {0, 0, 0, 0, 0, 0, 0, 0};
      // 6-arg form (clang-23 / therock-10.0): (g0, g1, g2, g3, g4, cpol)
      __builtin_amdgcn_tensor_load_to_lds(g0, g1, gz4, gz4, gz8, 0);
    }
    // --- Stage B: 32x128, transposed into Blds[n][k]; prefetch next k-step ---
    if (k0 + 32 < K)
      __builtin_prefetch(W + (size_t)(k0 + 32 + bk0) * N + bn + bn0, 0, 0);
#pragma unroll
    for (int kk = 0; kk < 2; ++kk) {
      v8bf w8 = *(const v8bf*)(W + (size_t)(k0 + bk0 + kk) * N + bn + bn0);
#pragma unroll
      for (int j = 0; j < 8; ++j) Blds[bn0 + j][bk0 + kk] = w8[j];
    }
    if (wave0) __builtin_amdgcn_s_wait_tensorcnt(0);
    __syncthreads();

    v16bf af = frag_ld(&Alds[wv * 16 + nlo][h * 8], 16);
#pragma unroll
    for (int nt = 0; nt < 8; ++nt) {
      v16bf bfr = frag_ld(&Blds[nt * 16 + nlo][h * 8], 16);
      acc[nt] = wmma_bf16(af, bfr, acc[nt]);
    }
    __syncthreads();
  }

  // Epilogue: C-layout element (VGPR r, lane) -> row = r + 8*h, col = nlo
#pragma unroll
  for (int nt = 0; nt < 8; ++nt) {
    const int col = bn + nt * 16 + nlo;
    const float bv = bias[col];
#pragma unroll
    for (int r = 0; r < 8; ++r) {
      const int row = bm + wv * 16 + r + 8 * h;
      float v = acc[nt][r] + bv;
      size_t idx = (size_t)row * N + col;
      if constexpr (EPI == EPI_BF16)           Cb[idx] = (bf16)v;
      else if constexpr (EPI == EPI_GELU_BF16) Cb[idx] = (bf16)gelu_new_f(v);
      else if constexpr (EPI == EPI_RES_F32)   Cf[idx] = R[idx] + v;
      else                                     Cf[idx] = v;
    }
  }
}

// Fused causal flash attention: one WG per (query block of 128, head).
// qkv: [T, 3*NX] bf16 ; o: [T, NX] bf16
__global__ void __launch_bounds__(256)
k_attn(const bf16* __restrict__ qkv, bf16* __restrict__ o) {
  __shared__ bf16 Qlds [128][72];   // [q][d]
  __shared__ bf16 Ktlds[128][72];   // [key][d]  (B-layout [n][k] for S gemm)
  __shared__ bf16 Vtlds[ 64][136];  // [d][key]  (B-layout [n][k] for PV gemm)
  __shared__ bf16 Plds [128][136];  // [q][key]  (A operand for PV gemm)

  const int qb = blockIdx.x;
  const int hh = blockIdx.y;
  const int t  = threadIdx.x;
  const int wv = t >> 5, lane = t & 31;
  const int h  = lane >> 4, nlo = lane & 15;

  { // load Q block
    const int r = t >> 1, dh = (t & 1) * 32;
    const bf16* gq = qkv + (size_t)(qb * 128 + r) * (3 * NXc) + hh * HDc + dh;
#pragma unroll
    for (int j = 0; j < 4; ++j)
      *(v8bf*)&Qlds[r][dh + j * 8] = *(const v8bf*)(gq + j * 8);
  }

  float m_i[8], l_i[8];
  v8f Ov[4] = {};
#pragma unroll
  for (int r = 0; r < 8; ++r) { m_i[r] = -1e30f; l_i[r] = 0.0f; }

  for (int kb = 0; kb <= qb; ++kb) {
    __syncthreads();
    { // load K, V block
      const int key = t >> 1, dh = (t & 1) * 32;
      const bf16* gk = qkv + (size_t)(kb * 128 + key) * (3 * NXc) + NXc     + hh * HDc + dh;
      const bf16* gv = qkv + (size_t)(kb * 128 + key) * (3 * NXc) + 2 * NXc + hh * HDc + dh;
#pragma unroll
      for (int j = 0; j < 4; ++j)
        *(v8bf*)&Ktlds[key][dh + j * 8] = *(const v8bf*)(gk + j * 8);
#pragma unroll
      for (int j = 0; j < 4; ++j) {
        v8bf vv = *(const v8bf*)(gv + j * 8);
#pragma unroll
        for (int e = 0; e < 8; ++e) Vtlds[dh + j * 8 + e][key] = vv[e];
      }
    }
    __syncthreads();

    // S = Q @ K^T  (wave: 16 q-rows x 128 keys), K-dim = 64
    v8f Sv[8] = {};
#pragma unroll
    for (int d0 = 0; d0 < 64; d0 += 32) {
      v16bf af = frag_ld(&Qlds[wv * 16 + nlo][d0 + h * 8], 16);
#pragma unroll
      for (int nt = 0; nt < 8; ++nt) {
        v16bf bfr = frag_ld(&Ktlds[nt * 16 + nlo][d0 + h * 8], 16);
        Sv[nt] = wmma_bf16(af, bfr, Sv[nt]);
      }
    }
    // scale + causal mask
#pragma unroll
    for (int nt = 0; nt < 8; ++nt) {
      const int col = kb * 128 + nt * 16 + nlo;
#pragma unroll
      for (int r = 0; r < 8; ++r) {
        const int row = qb * 128 + wv * 16 + r + 8 * h;
        float s = Sv[nt][r] * 0.125f; // 1/sqrt(64)
        Sv[nt][r] = (col <= row) ? s : -1e30f;
      }
    }
    // online softmax per row (rows r / r+8 live in lane halves -> width-16 shuffles)
#pragma unroll
    for (int r = 0; r < 8; ++r) {
      float mx = Sv[0][r];
#pragma unroll
      for (int nt = 1; nt < 8; ++nt) mx = fmaxf(mx, Sv[nt][r]);
#pragma unroll
      for (int s = 1; s < 16; s <<= 1) mx = fmaxf(mx, __shfl_xor(mx, s, 16));
      float mnew  = fmaxf(m_i[r], mx);
      float alpha = __expf(m_i[r] - mnew);
      m_i[r] = mnew;
      l_i[r] *= alpha;
#pragma unroll
      for (int dt = 0; dt < 4; ++dt) Ov[dt][r] *= alpha;
      float rs = 0.0f;
#pragma unroll
      for (int nt = 0; nt < 8; ++nt) {
        float p = __expf(Sv[nt][r] - mnew);
        Sv[nt][r] = p;
        rs += p;
      }
#pragma unroll
      for (int s = 1; s < 16; s <<= 1) rs += __shfl_xor(rs, s, 16);
      l_i[r] += rs;
    }
    // write P (C-layout -> LDS) then re-read as A operand
#pragma unroll
    for (int nt = 0; nt < 8; ++nt)
#pragma unroll
      for (int r = 0; r < 8; ++r)
        Plds[wv * 16 + r + 8 * h][nt * 16 + nlo] = (bf16)Sv[nt][r];
    __syncthreads();

    // O += P @ V  (K-dim 128)
#pragma unroll
    for (int kk0 = 0; kk0 < 128; kk0 += 32) {
      v16bf af = frag_ld(&Plds[wv * 16 + nlo][kk0 + h * 8], 16);
#pragma unroll
      for (int dt = 0; dt < 4; ++dt) {
        v16bf bfr = frag_ld(&Vtlds[dt * 16 + nlo][kk0 + h * 8], 16);
        Ov[dt] = wmma_bf16(af, bfr, Ov[dt]);
      }
    }
  }

  // normalize + store (heads re-interleaved into [T, NX])
#pragma unroll
  for (int dt = 0; dt < 4; ++dt)
#pragma unroll
    for (int r = 0; r < 8; ++r) {
      const int row = qb * 128 + wv * 16 + r + 8 * h;
      const int col = hh * HDc + dt * 16 + nlo;
      o[(size_t)row * NXc + col] = (bf16)(Ov[dt][r] / l_i[r]);
    }
}

// LayerNorm of one 1024-wide row per WG; fp32 in, bf16 out.
__global__ void __launch_bounds__(256)
k_layernorm(const float* __restrict__ x, const float* __restrict__ g,
            const float* __restrict__ b, bf16* __restrict__ out) {
  const int row = blockIdx.x;
  const int t = threadIdx.x;
  __shared__ float red[2][8];
  const float* xr = x + (size_t)row * NXc;
  float v[4], s = 0.f, s2 = 0.f;
#pragma unroll
  for (int i = 0; i < 4; ++i) {
    v[i] = xr[t + i * 256];
    s += v[i]; s2 += v[i] * v[i];
  }
#pragma unroll
  for (int sh = 1; sh < 32; sh <<= 1) {
    s  += __shfl_xor(s,  sh, 32);
    s2 += __shfl_xor(s2, sh, 32);
  }
  if ((t & 31) == 0) { red[0][t >> 5] = s; red[1][t >> 5] = s2; }
  __syncthreads();
  s = 0.f; s2 = 0.f;
#pragma unroll
  for (int i = 0; i < 8; ++i) { s += red[0][i]; s2 += red[1][i]; }
  const float mean = s * (1.0f / NXc);
  const float var  = s2 * (1.0f / NXc) - mean * mean;
  const float rstd = rsqrtf(var + 1e-5f);
#pragma unroll
  for (int i = 0; i < 4; ++i) {
    const int c = t + i * 256;
    out[(size_t)row * NXc + c] = (bf16)((v[i] - mean) * rstd * g[c] + b[c]);
  }
}

__global__ void k_add_wpe(const float* __restrict__ x, const float* __restrict__ wpe,
                          float* __restrict__ hbuf, int n) {
  int i = blockIdx.x * blockDim.x + threadIdx.x;
  if (i < n) hbuf[i] = x[i] + wpe[i];
}

__global__ void k_f32_to_bf16(const float* __restrict__ s, bf16* __restrict__ d, size_t n) {
  size_t i = (size_t)blockIdx.x * blockDim.x + threadIdx.x;
  if (i < n) d[i] = (bf16)s[i];
}

extern "C" void kernel_launch(void* const* d_in, const int* in_sizes, int n_in,
                              void* d_out, int out_size, void* d_ws, size_t ws_size,
                              hipStream_t stream) {
  (void)in_sizes; (void)n_in; (void)out_size; (void)ws_size;

  const float* x      = (const float*)d_in[0];
  const float* wpe    = (const float*)d_in[1];
  const float* ln1_g  = (const float*)d_in[2];
  const float* ln1_b  = (const float*)d_in[3];
  const float* attn_w = (const float*)d_in[4];
  const float* attn_b = (const float*)d_in[5];
  const float* proj_w = (const float*)d_in[6];
  const float* proj_b = (const float*)d_in[7];
  const float* ln2_g  = (const float*)d_in[8];
  const float* ln2_b  = (const float*)d_in[9];
  const float* fc_w   = (const float*)d_in[10];
  const float* fc_b   = (const float*)d_in[11];
  const float* fc2_w  = (const float*)d_in[12];
  const float* fc2_b  = (const float*)d_in[13];
  const float* lnf_g  = (const float*)d_in[14];
  const float* lnf_b  = (const float*)d_in[15];
  const float* mlpf_w = (const float*)d_in[16];
  const float* mlpf_b = (const float*)d_in[17];

  char* ws = (char*)d_ws;
  size_t off = 0;
  auto alloc = [&](size_t bytes) -> void* {
    void* p = ws + off;
    off += (bytes + 255) & ~(size_t)255;
    return p;
  };

  float* h    = (float*)alloc((size_t)Tt * NXc * 4);
  bf16*  xn   = (bf16*)alloc((size_t)Tt * NXc * 2);
  bf16*  qkvb = (bf16*)alloc((size_t)Tt * 3 * NXc * 2);
  bf16*  ob   = (bf16*)alloc((size_t)Tt * NXc * 2);
  bf16*  fca  = (bf16*)alloc((size_t)Tt * 4 * NXc * 2);
  bf16*  wA   = (bf16*)alloc((size_t)NLc * NXc * 3 * NXc * 2);
  bf16*  wP   = (bf16*)alloc((size_t)NLc * NXc * NXc * 2);
  bf16*  wF   = (bf16*)alloc((size_t)NLc * NXc * 4 * NXc * 2);
  bf16*  wF2  = (bf16*)alloc((size_t)NLc * 4 * NXc * NXc * 2);
  bf16*  wM   = (bf16*)alloc((size_t)NXc * NXc * 2);

  auto cvt = [&](const float* s, bf16* d, size_t n) {
    k_f32_to_bf16<<<dim3((unsigned)((n + 255) / 256)), 256, 0, stream>>>(s, d, n);
  };
  cvt(attn_w, wA,  (size_t)NLc * NXc * 3 * NXc);
  cvt(proj_w, wP,  (size_t)NLc * NXc * NXc);
  cvt(fc_w,   wF,  (size_t)NLc * NXc * 4 * NXc);
  cvt(fc2_w,  wF2, (size_t)NLc * 4 * NXc * NXc);
  cvt(mlpf_w, wM,  (size_t)NXc * NXc);

  k_add_wpe<<<dim3(Tt * NXc / 256), 256, 0, stream>>>(x, wpe, h, Tt * NXc);

  for (int l = 0; l < NLc; ++l) {
    k_layernorm<<<Tt, 256, 0, stream>>>(h, ln1_g + (size_t)l * NXc, ln1_b + (size_t)l * NXc, xn);

    k_gemm<EPI_BF16><<<dim3(3 * NXc / 128, Tt / 128), 256, 0, stream>>>(
        xn, wA + (size_t)l * NXc * 3 * NXc, attn_b + (size_t)l * 3 * NXc,
        nullptr, nullptr, qkvb, Tt, 3 * NXc, NXc);

    k_attn<<<dim3(Tt / 128, NHc), 256, 0, stream>>>(qkvb, ob);

    k_gemm<EPI_RES_F32><<<dim3(NXc / 128, Tt / 128), 256, 0, stream>>>(
        ob, wP + (size_t)l * NXc * NXc, proj_b + (size_t)l * NXc,
        h, h, nullptr, Tt, NXc, NXc);

    k_layernorm<<<Tt, 256, 0, stream>>>(h, ln2_g + (size_t)l * NXc, ln2_b + (size_t)l * NXc, xn);

    k_gemm<EPI_GELU_BF16><<<dim3(4 * NXc / 128, Tt / 128), 256, 0, stream>>>(
        xn, wF + (size_t)l * NXc * 4 * NXc, fc_b + (size_t)l * 4 * NXc,
        nullptr, nullptr, fca, Tt, 4 * NXc, NXc);

    k_gemm<EPI_RES_F32><<<dim3(NXc / 128, Tt / 128), 256, 0, stream>>>(
        fca, wF2 + (size_t)l * 4 * NXc * NXc, fc2_b + (size_t)l * NXc,
        h, h, nullptr, Tt, NXc, 4 * NXc);
  }

  k_layernorm<<<Tt, 256, 0, stream>>>(h, lnf_g, lnf_b, xn);

  k_gemm<EPI_F32><<<dim3(NXc / 128, Tt / 128), 256, 0, stream>>>(
      xn, wM, mlpf_b, nullptr, (float*)d_out, nullptr, Tt, NXc, NXc);
}
